// GraphConvolution_3659312136315
// MI455X (gfx1250) — matrix operand order repaired and checked by reading, not compiled, not run on hardware
//
#include <hip/hip_runtime.h>
#include <hip/hip_bf16.h>
#include <math.h>

typedef __bf16 bf16;
typedef __attribute__((ext_vector_type(16))) __bf16 v16bf;
typedef __attribute__((ext_vector_type(8)))  __bf16 v8bf;
typedef __attribute__((ext_vector_type(8)))  float  v8f;

#define HDIM 256

__device__ __forceinline__ float gelu_exact(float x) {
    // torch nn.GELU() default (erf variant)
    return 0.5f * x * (1.0f + erff(x * 0.70710678118654752f));
}

__device__ __forceinline__ v16bf load_afrag(const bf16* arow, int kb, int halfid) {
    // A 16x32 bf16 fragment: lanes 0-15 hold K kb+[0..15];
    // lanes 16-31 hold K kb+16+[0..15].
    v8bf alo = *(const v8bf*)(arow + kb + halfid * 16);
    v8bf ahi = *(const v8bf*)(arow + kb + halfid * 16 + 8);
    v16bf a;
#pragma unroll
    for (int i = 0; i < 8; ++i) { a[i] = alo[i]; a[i + 8] = ahi[i]; }
    return a;
}

// Wt[n*K + k] = (bf16) W[k*Ncols + n]   (col-major bf16 weights)
__global__ void k_transpose_bf16(const float* __restrict__ W, bf16* __restrict__ Wt,
                                 int K, int Ncols) {
    int i = blockIdx.x * blockDim.x + threadIdx.x;
    if (i >= K * Ncols) return;
    int n = i / K;
    int k = i - n * K;
    Wt[i] = (bf16)W[k * Ncols + n];
}

__global__ void k_cvt_bf16(const float* __restrict__ X, bf16* __restrict__ Y, int n) {
    int i = blockIdx.x * blockDim.x + threadIdx.x;
    if (i < n) Y[i] = (bf16)X[i];
}

// C = gelu(A[M,K](bf16) @ Wt[256,K]^T(bf16, col-major) + bias) (+ residual)
// 8 waves per block; each wave computes a 32x64 tile via 8x v_wmma_f32_16x16x32_bf16
// per K-step (two row-tiles share the four B fragments: 12 b128 loads / 8 WMMAs).
__global__ __launch_bounds__(256)
void k_gemm_bias_gelu(const bf16* __restrict__ A, const bf16* __restrict__ Wt,
                      const float* __restrict__ bias, int M, int K,
                      float* __restrict__ outf, const float* __restrict__ residual,
                      bf16* __restrict__ outb) {
    const int wave   = threadIdx.x >> 5;
    const int lane   = threadIdx.x & 31;
    const int lrow   = lane & 15;
    const int halfid = lane >> 4;

    const int Mtiles    = M >> 4;                    // M divisible by 16
    const int row_tile0 = (blockIdx.x * 8 + wave) * 2;
    if (row_tile0 >= Mtiles) return;                 // wave-uniform: EXEC all-ones
    const bool has2 = (row_tile0 + 1) < Mtiles;
    const int r0 = row_tile0 * 16;
    const int r1 = has2 ? r0 + 16 : r0;              // clamp: recompute tile0, skip store
    const int c0 = blockIdx.y * 64;

    v8f acc0[4], acc1[4];
#pragma unroll
    for (int t = 0; t < 4; ++t)
#pragma unroll
        for (int r = 0; r < 8; ++r) { acc0[t][r] = 0.0f; acc1[t][r] = 0.0f; }

    const bf16* arow0 = A + (size_t)(r0 + lrow) * K;
    const bf16* arow1 = A + (size_t)(r1 + lrow) * K;

    for (int kb = 0; kb < K; kb += 32) {
        v16bf a0 = load_afrag(arow0, kb, halfid);
        v16bf a1 = load_afrag(arow1, kb, halfid);
#pragma unroll
        for (int t = 0; t < 4; ++t) {
            // B 32x16: lane holds column (c0+t*16+lrow), K = kb + halfid*16 + [0..15],
            // contiguous in the transposed weight layout -> single 32B load.
            v16bf bfrag = *(const v16bf*)(Wt + (size_t)(c0 + t * 16 + lrow) * K
                                             + kb + halfid * 16);
            acc0[t] = __builtin_amdgcn_wmma_f32_16x16x32_bf16(
                false, a0, false, bfrag, (short)0, acc0[t], false, false);
            acc1[t] = __builtin_amdgcn_wmma_f32_16x16x32_bf16(
                false, a1, false, bfrag, (short)0, acc1[t], false, false);
        }
    }

    // Epilogue: bias + GELU (+ residual), dual f32/bf16 stores.
#pragma unroll
    for (int rt = 0; rt < 2; ++rt) {
        if (rt == 1 && !has2) break;
        const int rb = r0 + rt * 16;
#pragma unroll
        for (int t = 0; t < 4; ++t) {
            const int col = c0 + t * 16 + lrow;
            const float bb = bias[col];
            const v8f& a = rt ? acc1[t] : acc0[t];
#pragma unroll
            for (int r = 0; r < 8; ++r) {
                const int row = rb + r + halfid * 8;
                const size_t idx = (size_t)row * HDIM + col;
                float v = gelu_exact(a[r] + bb);
                if (residual) v += residual[idx];
                if (outf) outf[idx] = v;
                if (outb) outb[idx] = (bf16)v;
            }
        }
    }
}

// One wave per edge: agg[dst] += msg[src]; cnt[dst] += 1
__global__ __launch_bounds__(256)
void k_scatter(const float* __restrict__ msg, const long long* __restrict__ ei,
               float* __restrict__ agg, float* __restrict__ cnt, int E) {
    const int e = blockIdx.x * 8 + (threadIdx.x >> 5);
    if (e >= E) return;
    const int lane = threadIdx.x & 31;
    const long long s = ei[e];
    const long long d = ei[(long long)E + e];
    const float4* mrow = (const float4*)(msg + s * HDIM);
    float* arow = agg + d * HDIM;
#pragma unroll
    for (int i = 0; i < 2; ++i) {
        const int q = lane + i * 32;          // 64 float4 per row
        float4 m = mrow[q];
        atomicAdd(arow + q * 4 + 0, m.x);
        atomicAdd(arow + q * 4 + 1, m.y);
        atomicAdd(arow + q * 4 + 2, m.z);
        atomicAdd(arow + q * 4 + 3, m.w);
    }
    if (lane == 0) atomicAdd(cnt + d, 1.0f);
}

// comb[n, 0:256]=target, comb[n, 256:512]=agg/max(cnt,1)   (bf16 for GEMM W2)
__global__ void k_finalize(const float* __restrict__ target, const float* __restrict__ agg,
                           const float* __restrict__ cnt, bf16* __restrict__ comb, int N) {
    int i = blockIdx.x * blockDim.x + threadIdx.x;
    if (i >= N * HDIM) return;
    int row = i >> 8;
    int col = i & 255;
    float inv = 1.0f / fmaxf(cnt[row], 1.0f);
    comb[(size_t)row * 512 + col]       = (bf16)target[i];
    comb[(size_t)row * 512 + 256 + col] = (bf16)(agg[i] * inv);
}

// Row-wise LayerNorm over 256 features; one 256-thread block per row.
__global__ __launch_bounds__(256)
void k_layernorm(const float* __restrict__ x, const float* __restrict__ gamma,
                 const float* __restrict__ beta, float* __restrict__ out) {
    __shared__ float sdata[256];
    const int row = blockIdx.x;
    const int tid = threadIdx.x;
    const float v = x[(size_t)row * HDIM + tid];
    sdata[tid] = v;
    __syncthreads();
    for (int s = 128; s > 0; s >>= 1) {
        if (tid < s) sdata[tid] += sdata[tid + s];
        __syncthreads();
    }
    const float mu = sdata[0] * (1.0f / HDIM);
    __syncthreads();
    const float d = v - mu;
    sdata[tid] = d * d;
    __syncthreads();
    for (int s = 128; s > 0; s >>= 1) {
        if (tid < s) sdata[tid] += sdata[tid + s];
        __syncthreads();
    }
    const float var = sdata[0] * (1.0f / HDIM);
    out[(size_t)row * HDIM + tid] = d * rsqrtf(var + 1e-5f) * gamma[tid] + beta[tid];
}

extern "C" void kernel_launch(void* const* d_in, const int* in_sizes, int n_in,
                              void* d_out, int out_size, void* d_ws, size_t ws_size,
                              hipStream_t stream) {
    const float*     source = (const float*)d_in[0];
    const float*     target = (const float*)d_in[1];
    const long long* ei     = (const long long*)d_in[2];
    const float* W0 = (const float*)d_in[4];  const float* b0 = (const float*)d_in[5];
    const float* W1 = (const float*)d_in[6];  const float* b1 = (const float*)d_in[7];
    const float* W2 = (const float*)d_in[8];  const float* b2 = (const float*)d_in[9];
    const float* W3 = (const float*)d_in[10]; const float* b3 = (const float*)d_in[11];
    const float* gamma = (const float*)d_in[12];
    const float* beta  = (const float*)d_in[13];

    const int N = in_sizes[0] / HDIM;   // 50000
    const int E = in_sizes[2] / 2;      // 320000

    // Workspace layout (256B aligned). Total ~231 MB.
    char* ws = (char*)d_ws;
    size_t off = 0;
    auto alloc = [&](size_t bytes) -> void* {
        void* p = ws + off;
        off += (bytes + 255) & ~(size_t)255;
        return p;
    };
    bf16*  srcbf = (bf16*)alloc((size_t)N * HDIM * sizeof(bf16));
    bf16*  h0bf  = (bf16*)alloc((size_t)N * HDIM * sizeof(bf16));
    bf16*  h2bf  = (bf16*)alloc((size_t)N * HDIM * sizeof(bf16));
    float* msg   = (float*)alloc((size_t)N * HDIM * sizeof(float)); // reused as pre-LN buf
    float* agg   = (float*)alloc((size_t)N * HDIM * sizeof(float));
    float* cnt   = (float*)alloc((size_t)N * sizeof(float));
    bf16*  comb  = (bf16*)alloc((size_t)N * 2 * HDIM * sizeof(bf16));
    bf16*  W0t   = (bf16*)alloc((size_t)HDIM * HDIM * sizeof(bf16));
    bf16*  W1t   = (bf16*)alloc((size_t)HDIM * HDIM * sizeof(bf16));
    bf16*  W2t   = (bf16*)alloc((size_t)2 * HDIM * HDIM * sizeof(bf16));
    bf16*  W3t   = (bf16*)alloc((size_t)HDIM * HDIM * sizeof(bf16));
    float* pre   = msg;

    // Zero accumulators (graph-capture-safe async memsets)
    hipMemsetAsync(agg, 0, (size_t)N * HDIM * sizeof(float), stream);
    hipMemsetAsync(cnt, 0, (size_t)N * sizeof(float), stream);

    // Weight prep: transpose + bf16
    k_transpose_bf16<<<(HDIM * HDIM + 255) / 256, 256, 0, stream>>>(W0, W0t, HDIM, HDIM);
    k_transpose_bf16<<<(HDIM * HDIM + 255) / 256, 256, 0, stream>>>(W1, W1t, HDIM, HDIM);
    k_transpose_bf16<<<(2 * HDIM * HDIM + 255) / 256, 256, 0, stream>>>(W2, W2t, 2 * HDIM, HDIM);
    k_transpose_bf16<<<(HDIM * HDIM + 255) / 256, 256, 0, stream>>>(W3, W3t, HDIM, HDIM);
    k_cvt_bf16<<<((size_t)N * HDIM + 255) / 256, 256, 0, stream>>>(source, srcbf, N * HDIM);

    const int Mtiles = N / 16;                       // 3125
    const int gx = (Mtiles + 15) / 16;               // 8 waves x 2 row-tiles per block
    dim3 ggrid(gx, 4);

    // msg MLP: h0 = gelu(src@W0+b0) ; msg = gelu(h0@W1+b1)
    k_gemm_bias_gelu<<<ggrid, 256, 0, stream>>>(srcbf, W0t, b0, N, HDIM,
                                                nullptr, nullptr, h0bf);
    k_gemm_bias_gelu<<<ggrid, 256, 0, stream>>>(h0bf, W1t, b1, N, HDIM,
                                                msg, nullptr, nullptr);

    // gather + scatter-mean
    k_scatter<<<(E + 7) / 8, 256, 0, stream>>>(msg, ei, agg, cnt, E);
    k_finalize<<<((size_t)N * HDIM + 255) / 256, 256, 0, stream>>>(target, agg, cnt, comb, N);

    // combine MLP: h2 = gelu(comb@W2+b2) ; pre = gelu(h2@W3+b3) + target
    k_gemm_bias_gelu<<<ggrid, 256, 0, stream>>>(comb, W2t, b2, N, 2 * HDIM,
                                                nullptr, nullptr, h2bf);
    k_gemm_bias_gelu<<<ggrid, 256, 0, stream>>>(h2bf, W3t, b3, N, HDIM,
                                                pre, target, nullptr);

    // LayerNorm -> output
    k_layernorm<<<N, 256, 0, stream>>>(pre, gamma, beta, (float*)d_out);
}
